// SuperGlue_18605798326439
// MI455X (gfx1250) — matrix operand order repaired and checked by reading, not compiled, not run on hardware
//
#include <hip/hip_runtime.h>
#include <hip/hip_bf16.h>
#include <math.h>

typedef __attribute__((ext_vector_type(16))) _Float16 v16h;
typedef __attribute__((ext_vector_type(8)))  float    v8f;

#define LDC       1032                    // padded leading dim of 1025x1025 coupling
#define NORMC    (-7.624618986159398f)    // -log(2048)
#define LOG2N    ( 7.624618986159398f)    // +log(2048)
#define LOG_HALF (-0.6931471805599453f)   // log(1024)-log(2048)

// LDS tile stride (floats): multiple of 4 (16B-aligned float4 rows) and
// 36 mod 64 banks walks all 16 lane-groups conflict-free.
#define TSTRIDE 36

static __device__ __forceinline__ unsigned lds_off(const void* p) {
  // generic pointer to __shared__ : low 32 bits are the LDS offset (ISA 10.2)
  return (unsigned)(size_t)p;
}

// ---------------------------------------------------------------------------
// Double-buffered LDS-staged WMMA GEMM:  Y = epilogue( scale * op(A) * op(B) )
// AMODE 0: A[row*lda+k] (async-to-LDS)    1: A[k*lda+row] (VGPR transpose stage)
// BMODE 0: B[k*ldb+col] (VGPR transpose)  1: B[col*ldb+k] (async-to-LDS)
// BMODE 2: concat rows <ksplit from B1, rest from B2 (VGPR transpose)
// Block = 256 threads = 8 waves (2x4) -> 32(M) x 128(N) macro-tile.
// Each wave: 16x32 strip = 2 x v_wmma_f32_16x16x32_f16 per 32-K slab,
// reusing one A fragment. Slab n+1 staged (async) while slab n computes.
// ---------------------------------------------------------------------------
template<int AMODE, int BMODE>
__global__ __launch_bounds__(256) void gemm_wmma(
    const float* __restrict__ A, int lda, long hsA,
    const float* __restrict__ B1, int ldb, long hsB,
    const float* __restrict__ B2, int ksplit,
    const float* __restrict__ bias,
    const float* __restrict__ bng, const float* __restrict__ bnb,
    const float* __restrict__ bnm, const float* __restrict__ bnv,
    const float* __restrict__ resid, int ldr,
    float* __restrict__ Y, int ldy, long hsY,
    int M, int N, int K, float scale)
{
  __shared__ __align__(16) float Al[2][32][TSTRIDE];
  __shared__ __align__(16) float Bl[2][128][TSTRIDE];
  const int t    = threadIdx.x;
  const int wv   = t >> 5;
  const int lane = t & 31;
  const int nMacroN = N >> 7;
  const int row0 = (blockIdx.x / nMacroN) << 5;   // 32 rows / macro
  const int col0 = (blockIdx.x % nMacroN) << 7;   // 128 cols / macro
  const float* Ah = A  + hsA * (long)blockIdx.z;
  const float* Bh = B1 + hsB * (long)blockIdx.z;
  float*       Yh = Y  + hsY * (long)blockIdx.z;
  const int wy = wv >> 2, wx = wv & 3;            // 2x4 waves
  const int lr = lane & 15;
  const int koffA = (lane >> 4) << 4;             // A: K contiguous koffA..koffA+15
  const int koffB = (lane >> 4) << 4;             // B: K contiguous koffB..koffB+15

  auto stageA = [&](int p, int k0) {
    if constexpr (AMODE == 0) {
      const int r = t >> 3, kq = (t & 7) << 2;    // one b128 per thread
      const unsigned la = lds_off(&Al[p][r][kq]);
      const float* ga = &Ah[(long)(row0 + r) * lda + k0 + kq];
      asm volatile("global_load_async_to_lds_b128 %0, %1, off"
                   :: "v"(la), "v"(ga) : "memory");
    } else {
      const int r = t & 31, kb = (t >> 5) << 2;   // coalesced over rows
#pragma unroll
      for (int q = 0; q < 4; ++q)
        Al[p][r][kb + q] = Ah[(long)(k0 + kb + q) * lda + row0 + r];
    }
  };
  auto stageB = [&](int p, int k0) {
    if constexpr (BMODE == 1) {
      const int c = t >> 1, kq = (t & 1) << 4;    // four b128 per thread
      const unsigned lb = lds_off(&Bl[p][c][kq]);
      const float* gb = &Bh[(long)(col0 + c) * ldb + k0 + kq];
      asm volatile("global_load_async_to_lds_b128 %0, %1, off\n\t"
                   "global_load_async_to_lds_b128 %2, %3, off\n\t"
                   "global_load_async_to_lds_b128 %4, %5, off\n\t"
                   "global_load_async_to_lds_b128 %6, %7, off"
                   :: "v"(lb),        "v"(gb),
                      "v"(lb + 16u),  "v"(gb + 4),
                      "v"(lb + 32u),  "v"(gb + 8),
                      "v"(lb + 48u),  "v"(gb + 12) : "memory");
    } else {
      const int c = t & 127, kb = (t >> 7) << 4;  // coalesced over cols, transpose in
#pragma unroll
      for (int q = 0; q < 16; ++q) {
        const int kg = k0 + kb + q;
        float x;
        if constexpr (BMODE == 0) x = Bh[(long)kg * ldb + col0 + c];
        else x = (kg < ksplit) ? Bh[(long)kg * ldb + col0 + c]
                               : B2[(long)(kg - ksplit) * ldb + col0 + c];
        Bl[p][c][kb + q] = x;
      }
    }
  };
  auto cvt16 = [](v16h& d, const float4& f0, const float4& f1,
                  const float4& f2, const float4& f3) {
    d[0]=(_Float16)f0.x;  d[1]=(_Float16)f0.y;  d[2]=(_Float16)f0.z;  d[3]=(_Float16)f0.w;
    d[4]=(_Float16)f1.x;  d[5]=(_Float16)f1.y;  d[6]=(_Float16)f1.z;  d[7]=(_Float16)f1.w;
    d[8]=(_Float16)f2.x;  d[9]=(_Float16)f2.y;  d[10]=(_Float16)f2.z; d[11]=(_Float16)f2.w;
    d[12]=(_Float16)f3.x; d[13]=(_Float16)f3.y; d[14]=(_Float16)f3.z; d[15]=(_Float16)f3.w;
  };

  v8f acc0 = {}, acc1 = {};
  stageA(0, 0); stageB(0, 0);
  if constexpr (AMODE == 0 || BMODE == 1)
    asm volatile("s_wait_asynccnt 0x0" ::: "memory");
  __syncthreads();

  int p = 0;
  for (int k0 = 32; k0 <= K; k0 += 32) {
    const bool more = (k0 < K);
    if (more) { stageA(p ^ 1, k0); stageB(p ^ 1, k0); }  // prefetch next slab

    // fragments for slab (k0-32) from buffer p
    v16h av, b0v, b1v;
    {
      const float4* pa = (const float4*)&Al[p][(wy << 4) + lr][koffA];
      cvt16(av, pa[0], pa[1], pa[2], pa[3]);
      const float4* pb0 = (const float4*)&Bl[p][(wx << 5) + lr][koffB];
      cvt16(b0v, pb0[0], pb0[1], pb0[2], pb0[3]);
      const float4* pb1 = (const float4*)&Bl[p][(wx << 5) + 16 + lr][koffB];
      cvt16(b1v, pb1[0], pb1[1], pb1[2], pb1[3]);
    }
    acc0 = __builtin_amdgcn_wmma_f32_16x16x32_f16(false, av, false, b0v,
                                                  (short)0, acc0, false, false);
    acc1 = __builtin_amdgcn_wmma_f32_16x16x32_f16(false, av, false, b1v,
                                                  (short)0, acc1, false, false);
    if (more) {
      if constexpr (AMODE == 0 || BMODE == 1)
        asm volatile("s_wait_asynccnt 0x0" ::: "memory");
      __syncthreads();
      p ^= 1;
    }
  }

  // f32 C/D layout: lanes 0-15 -> M=j, lanes 16-31 -> M=8+j ; N = lane&15
  const int mbase = row0 + (wy << 4) + ((lane >> 4) << 3);
  const int nc0   = col0 + (wx << 5) + lr;
#pragma unroll
  for (int j = 0; j < 8; ++j) {
    const int m = mbase + j;
    float v0 = acc0[j] * scale;
    float v1 = acc1[j] * scale;
    if (bias) { const float b = bias[m]; v0 += b; v1 += b; }
    if (bng) {
      const float s = bng[m] * rsqrtf(bnv[m] + 1e-5f);
      v0 = fmaxf((v0 - bnm[m]) * s + bnb[m], 0.f);
      v1 = fmaxf((v1 - bnm[m]) * s + bnb[m], 0.f);
    }
    if (resid) {
      v0 += resid[(long)m * ldr + nc0];
      v1 += resid[(long)m * ldr + nc0 + 16];
    }
    Yh[(long)m * ldy + nc0]      = v0;
    Yh[(long)m * ldy + nc0 + 16] = v1;
  }
}

// ---------------------------------------------------------------------------
// Keypoint encoder MLP [3,32,64,128,256] + residual add of descriptors.
// ---------------------------------------------------------------------------
struct KencP {
  const float* w[4]; const float* b[4];
  const float* g[3]; const float* be[3]; const float* mn[3]; const float* vr[3];
};

__global__ __launch_bounds__(256) void kenc_kernel(
    const float* __restrict__ desc, const float* __restrict__ kpts,
    const float* __restrict__ scr, KencP P, float* __restrict__ out)
{
  __shared__ float a[256], bbuf[256];
  const int p = blockIdx.x, t = threadIdx.x;
  if (t == 0) a[0] = (kpts[p * 2 + 0] - 320.f) * (1.f / 448.f);
  if (t == 1) a[1] = (kpts[p * 2 + 1] - 240.f) * (1.f / 448.f);
  if (t == 2) a[2] = scr[p];
  __syncthreads();
  if (t < 32) {
    float s = P.b[0][t];
    for (int i = 0; i < 3; ++i) s += P.w[0][t * 3 + i] * a[i];
    const float sc = P.g[0][t] * rsqrtf(P.vr[0][t] + 1e-5f);
    bbuf[t] = fmaxf((s - P.mn[0][t]) * sc + P.be[0][t], 0.f);
  }
  __syncthreads();
  if (t < 64) {
    float s = P.b[1][t];
    for (int i = 0; i < 32; ++i) s += P.w[1][t * 32 + i] * bbuf[i];
    const float sc = P.g[1][t] * rsqrtf(P.vr[1][t] + 1e-5f);
    a[t] = fmaxf((s - P.mn[1][t]) * sc + P.be[1][t], 0.f);
  }
  __syncthreads();
  if (t < 128) {
    float s = P.b[2][t];
    for (int i = 0; i < 64; ++i) s += P.w[2][t * 64 + i] * a[i];
    const float sc = P.g[2][t] * rsqrtf(P.vr[2][t] + 1e-5f);
    bbuf[t] = fmaxf((s - P.mn[2][t]) * sc + P.be[2][t], 0.f);
  }
  __syncthreads();
  {
    float s = P.b[3][t];
    for (int i = 0; i < 128; ++i) s += P.w[3][t * 128 + i] * bbuf[i];
    out[(long)t * 1024 + p] = desc[(long)t * 1024 + p] + s;
  }
}

// ---------------------------------------------------------------------------
// Row softmax over attention scores S[4][1024][1024]; one block per row.
// ---------------------------------------------------------------------------
__global__ __launch_bounds__(256) void softmax_rows(float* __restrict__ S, int ncols)
{
  float* p = S + (long)blockIdx.x * ncols;
  __shared__ float red[256];
  float m = -1e30f;
  for (int j = threadIdx.x; j < ncols; j += 256) m = fmaxf(m, p[j]);
  red[threadIdx.x] = m; __syncthreads();
  for (int s = 128; s > 0; s >>= 1) {
    if (threadIdx.x < s) red[threadIdx.x] = fmaxf(red[threadIdx.x], red[threadIdx.x + s]);
    __syncthreads();
  }
  m = red[0]; __syncthreads();
  float sum = 0.f;
  for (int j = threadIdx.x; j < ncols; j += 256) {
    const float e = __expf(p[j] - m); p[j] = e; sum += e;
  }
  red[threadIdx.x] = sum; __syncthreads();
  for (int s = 128; s > 0; s >>= 1) {
    if (threadIdx.x < s) red[threadIdx.x] += red[threadIdx.x + s];
    __syncthreads();
  }
  const float inv = 1.f / red[0];
  for (int j = threadIdx.x; j < ncols; j += 256) p[j] *= inv;
}

// ---------------------------------------------------------------------------
// Sinkhorn machinery on the 1025x1025 coupling (leading dim LDC).
// ---------------------------------------------------------------------------
__global__ void build_coup(const float* __restrict__ sc, const float* __restrict__ alpha,
                           float* __restrict__ C)
{
  const int j = blockIdx.x * 256 + threadIdx.x;
  const int i = blockIdx.y;
  if (j > 1024) return;
  C[(long)i * LDC + j] = (i < 1024 && j < 1024) ? sc[(long)i * 1024 + j] : alpha[0];
}

__global__ void zero_uv(float* __restrict__ u, float* __restrict__ v)
{
  const int i = blockIdx.x * 256 + threadIdx.x;
  if (i < 1056) { u[i] = 0.f; v[i] = 0.f; }
}

__global__ __launch_bounds__(256) void sink_row(const float* __restrict__ C,
                                                const float* __restrict__ vv,
                                                float* __restrict__ uu)
{
  const int i = blockIdx.x;                   // 0..1024
  const float* row = C + (long)i * LDC;
  float m = -1e30f, s = 0.f;
  for (int j = threadIdx.x; j <= 1024; j += 256) {
    const float x = row[j] + vv[j];
    if (x > m) { s *= __expf(m - x); m = x; }
    s += __expf(x - m);
  }
  __shared__ float sm[256], ss[256];
  sm[threadIdx.x] = m; ss[threadIdx.x] = s; __syncthreads();
  for (int t = 128; t > 0; t >>= 1) {
    if (threadIdx.x < t) {
      const float m1 = sm[threadIdx.x], s1 = ss[threadIdx.x];
      const float m2 = sm[threadIdx.x + t], s2 = ss[threadIdx.x + t];
      const float M = fmaxf(m1, m2);
      sm[threadIdx.x] = M;
      ss[threadIdx.x] = s1 * __expf(m1 - M) + s2 * __expf(m2 - M);
    }
    __syncthreads();
  }
  if (threadIdx.x == 0) {
    const float lm = (i < 1024) ? NORMC : LOG_HALF;
    uu[i] = lm - (sm[0] + __logf(ss[0]));
  }
}

__global__ __launch_bounds__(256) void sink_col(const float* __restrict__ C,
                                                const float* __restrict__ uu,
                                                float* __restrict__ vv)
{
  const int col  = blockIdx.x * 32 + (threadIdx.x & 31);
  const int part = threadIdx.x >> 5;          // 0..7 row chunks
  float m = -1e30f, s = 0.f;
  if (col <= 1024) {
    for (int i = part; i <= 1024; i += 8) {
      const float x = C[(long)i * LDC + col] + uu[i];
      if (x > m) { s *= __expf(m - x); m = x; }
      s += __expf(x - m);
    }
  }
  __shared__ float sm[256], ss[256];
  sm[threadIdx.x] = m; ss[threadIdx.x] = s; __syncthreads();
  if (part == 0 && col <= 1024) {
    float M = m, S = s;
    for (int q = 1; q < 8; ++q) {
      const float m2 = sm[q * 32 + (threadIdx.x & 31)];
      const float s2 = ss[q * 32 + (threadIdx.x & 31)];
      if (m2 > M) { S = S * __expf(M - m2) + s2; M = m2; }
      else          S += s2 * __expf(m2 - M);
    }
    const float ln = (col < 1024) ? NORMC : LOG_HALF;
    vv[col] = ln - (M + __logf(S));
  }
}

// ---------------------------------------------------------------------------
// Matching: row/col argmax of Z = C + u + v + log(2048), mutual check.
// ---------------------------------------------------------------------------
__global__ __launch_bounds__(256) void row_argmax(const float* __restrict__ C,
                                                  const float* __restrict__ u,
                                                  const float* __restrict__ v,
                                                  float* __restrict__ mx, int* __restrict__ ix)
{
  const int i = blockIdx.x;
  float best = -1e30f; int bj = 0;
  for (int j = threadIdx.x; j < 1024; j += 256) {
    const float x = C[(long)i * LDC + j] + u[i] + v[j] + LOG2N;
    if (x > best) { best = x; bj = j; }
  }
  __shared__ float bm[256]; __shared__ int bi[256];
  bm[threadIdx.x] = best; bi[threadIdx.x] = bj; __syncthreads();
  for (int s = 128; s > 0; s >>= 1) {
    if (threadIdx.x < s && bm[threadIdx.x + s] > bm[threadIdx.x]) {
      bm[threadIdx.x] = bm[threadIdx.x + s]; bi[threadIdx.x] = bi[threadIdx.x + s];
    }
    __syncthreads();
  }
  if (threadIdx.x == 0) { mx[i] = bm[0]; ix[i] = bi[0]; }
}

__global__ void col_argmax(const float* __restrict__ C, const float* __restrict__ u,
                           const float* __restrict__ v, float* __restrict__ mx,
                           int* __restrict__ ix)
{
  const int j = blockIdx.x * 256 + threadIdx.x;
  if (j >= 1024) return;
  float best = -1e30f; int bi = 0;
  for (int i = 0; i < 1024; ++i) {
    const float x = C[(long)i * LDC + j] + u[i] + v[j] + LOG2N;
    if (x > best) { best = x; bi = i; }
  }
  mx[j] = best; ix[j] = bi;
}

__global__ void match0_kernel(const float* __restrict__ max0, const int* __restrict__ idx0,
                              const int* __restrict__ idx1,
                              int* __restrict__ matches0, float* __restrict__ ms0)
{
  const int i = blockIdx.x * 256 + threadIdx.x;
  if (i >= 1024) return;
  const int j0 = idx0[i];
  const bool mutual = (idx1[j0] == i);
  const float sc = mutual ? __expf(max0[i]) : 0.f;
  ms0[i] = sc;
  matches0[i] = (mutual && sc > 0.2f) ? j0 : -1;
}

__global__ void match1_kernel(const int* __restrict__ idx0, const int* __restrict__ idx1,
                              const int* __restrict__ matches0, const float* __restrict__ ms0,
                              int* __restrict__ matches1, float* __restrict__ ms1)
{
  const int j = blockIdx.x * 256 + threadIdx.x;
  if (j >= 1024) return;
  const int i1 = idx1[j];
  const bool mutual = (idx0[i1] == j);
  ms1[j] = mutual ? ms0[i1] : 0.f;
  matches1[j] = (mutual && matches0[i1] >= 0) ? i1 : -1;
}

// ---------------------------------------------------------------------------
// Host orchestration
// ---------------------------------------------------------------------------
static inline int gblk(int M, int N) { return (M >> 5) * (N >> 7); }  // 32x128 macros

extern "C" void kernel_launch(void* const* d_in, const int* in_sizes, int n_in,
                              void* d_out, int out_size, void* d_ws, size_t ws_size,
                              hipStream_t stream)
{
  (void)in_sizes; (void)out_size; (void)ws_size;
  auto F = [&](int i) -> const float* {
    const int ci = (i < n_in) ? i : (n_in > 0 ? n_in - 1 : 0);
    return (const float*)d_in[ci];
  };
  const float* desc0 = F(0); const float* desc1 = F(1);
  const float* kpts0 = F(2); const float* kpts1 = F(3);
  const float* scr0  = F(4); const float* scr1  = F(5);
  const float* alpha = F(6);
  const float* finb  = F(7); const float* finw = F(8);
  const int KENC_BASE = 9 + 18 * 16;  // 297

  float* ws = (float*)d_ws;
  size_t o = 0;
  auto carve = [&](size_t n) { float* p = ws + o; o += n; return p; };
  float* d0    = carve(262144);
  float* d1    = carve(262144);
  float* qbuf  = carve(262144);
  float* kbuf  = carve(262144);
  float* vbuf  = carve(262144);
  float* msg   = carve(262144);
  float* hid   = carve(524288);
  float* S     = carve(4u * 1048576u);
  float* sc    = carve(1048576);
  float* coup  = carve((size_t)1025 * LDC);
  float* u     = carve(1056);
  float* v     = carve(1056);
  float* dl0   = carve(262144);
  float* dl1   = carve(262144);
  float* max0  = carve(1024);
  float* max1  = carve(1024);
  int*   idx0  = (int*)carve(1024);
  int*   idx1  = (int*)carve(1024);

  const dim3 blk(256);

  // ---- keypoint encoder ----
  KencP kp;
  for (int st = 0; st < 3; ++st) {
    const int b = KENC_BASE + st * 6;
    kp.b[st]  = F(b + 0); kp.be[st] = F(b + 1); kp.g[st] = F(b + 2);
    kp.mn[st] = F(b + 3); kp.vr[st] = F(b + 4); kp.w[st] = F(b + 5);
  }
  kp.b[3] = F(KENC_BASE + 18); kp.w[3] = F(KENC_BASE + 19);
  kenc_kernel<<<1024, blk, 0, stream>>>(desc0, kpts0, scr0, kp, d0);
  kenc_kernel<<<1024, blk, 0, stream>>>(desc1, kpts1, scr1, kp, d1);

  // ---- 18 GNN layers ----
  const float* NP = nullptr;
  for (int L = 0; L < 18; ++L) {
    const bool cross = (L & 1);
    const int base = 9 + L * 16;
    const float *kb_ = F(base + 0),  *kw  = F(base + 1);
    const float *mb_ = F(base + 2),  *mw  = F(base + 3);
    const float *m0b = F(base + 4),  *m0be= F(base + 5), *m0g = F(base + 6);
    const float *m0m = F(base + 7),  *m0v = F(base + 8), *m0w = F(base + 9);
    const float *m1b = F(base + 10), *m1w = F(base + 11);
    const float *qb_ = F(base + 12), *qw  = F(base + 13);
    const float *vb_ = F(base + 14), *vw  = F(base + 15);
    float* xs[2] = {d0, d1};
    float* dls[2] = {dl0, dl1};
    for (int sI = 0; sI < 2; ++sI) {
      const float* x   = xs[sI];
      const float* src = cross ? xs[1 - sI] : xs[sI];
      gemm_wmma<0,0><<<dim3(gblk(256,1024)), blk, 0, stream>>>(
          qw,256,0L, x,1024,0L, NP,0, qb_, NP,NP,NP,NP, NP,0,
          qbuf,1024,0L, 256,1024,256, 1.f);
      gemm_wmma<0,0><<<dim3(gblk(256,1024)), blk, 0, stream>>>(
          kw,256,0L, src,1024,0L, NP,0, kb_, NP,NP,NP,NP, NP,0,
          kbuf,1024,0L, 256,1024,256, 1.f);
      gemm_wmma<0,0><<<dim3(gblk(256,1024)), blk, 0, stream>>>(
          vw,256,0L, src,1024,0L, NP,0, vb_, NP,NP,NP,NP, NP,0,
          vbuf,1024,0L, 256,1024,256, 1.f);
      // scores S[h] = (Q_h^T K_h)/8 : M=N=1024, K=64, head = blockIdx.z
      gemm_wmma<1,0><<<dim3(gblk(1024,1024),1,4), blk, 0, stream>>>(
          qbuf,4096,1024L, kbuf,4096,1024L, NP,0, NP, NP,NP,NP,NP, NP,0,
          S,1024,1048576L, 1024,1024,64, 0.125f);
      softmax_rows<<<4096, blk, 0, stream>>>(S, 1024);
      // msg_h = V_h * P_h^T : M=64, N=1024, K=1024 (async B path)
      gemm_wmma<0,1><<<dim3(gblk(64,1024),1,4), blk, 0, stream>>>(
          vbuf,4096,1024L, S,1024,1048576L, NP,0, NP, NP,NP,NP,NP, NP,0,
          msg,4096,1024L, 64,1024,1024, 1.f);
      gemm_wmma<0,0><<<dim3(gblk(256,1024)), blk, 0, stream>>>(
          mw,256,0L, msg,1024,0L, NP,0, mb_, NP,NP,NP,NP, NP,0,
          qbuf,1024,0L, 256,1024,256, 1.f);
      gemm_wmma<0,2><<<dim3(gblk(512,1024)), blk, 0, stream>>>(
          m0w,512,0L, x,1024,0L, qbuf,256, m0b, m0g,m0be,m0m,m0v, NP,0,
          hid,1024,0L, 512,1024,512, 1.f);
      gemm_wmma<0,0><<<dim3(gblk(256,1024)), blk, 0, stream>>>(
          m1w,512,0L, hid,1024,0L, NP,0, m1b, NP,NP,NP,NP, x,1024,
          dls[sI],1024,0L, 256,1024,512, 1.f);
    }
    { float* t = d0; d0 = dl0; dl0 = t; }
    { float* t = d1; d1 = dl1; dl1 = t; }
  }

  // ---- final projection + score matrix ----
  gemm_wmma<0,0><<<dim3(gblk(256,1024)), blk, 0, stream>>>(
      finw,256,0L, d0,1024,0L, NP,0, finb, NP,NP,NP,NP, NP,0,
      qbuf,1024,0L, 256,1024,256, 1.f);
  gemm_wmma<0,0><<<dim3(gblk(256,1024)), blk, 0, stream>>>(
      finw,256,0L, d1,1024,0L, NP,0, finb, NP,NP,NP,NP, NP,0,
      kbuf,1024,0L, 256,1024,256, 1.f);
  gemm_wmma<1,0><<<dim3(gblk(1024,1024)), blk, 0, stream>>>(
      qbuf,1024,0L, kbuf,1024,0L, NP,0, NP, NP,NP,NP,NP, NP,0,
      sc,1024,0L, 1024,1024,256, 0.0625f);

  // ---- Sinkhorn (100 iterations) ----
  build_coup<<<dim3(5, 1025), blk, 0, stream>>>(sc, alpha, coup);
  zero_uv<<<5, blk, 0, stream>>>(u, v);
  for (int it = 0; it < 100; ++it) {
    sink_row<<<1025, blk, 0, stream>>>(coup, v, u);
    sink_col<<<33,  blk, 0, stream>>>(coup, u, v);
  }

  // ---- matching ----
  row_argmax<<<1024, blk, 0, stream>>>(coup, u, v, max0, idx0);
  col_argmax<<<4,    blk, 0, stream>>>(coup, u, v, max1, idx1);
  int*   mo = (int*)d_out;
  float* fo = (float*)d_out;
  match0_kernel<<<4, blk, 0, stream>>>(max0, idx0, idx1, mo, fo + 2048);
  match1_kernel<<<4, blk, 0, stream>>>(idx0, idx1, mo, fo + 2048, mo + 1024, fo + 3072);
}
